// Qwen3InfAttention_5248450036531
// MI455X (gfx1250) — matrix operand order repaired and checked by reading, not compiled, not run on hardware
//
#include <hip/hip_runtime.h>

// ---------------------------------------------------------------------------
// Qwen3-style attention block for gfx1250 (MI455X), bf16 WMMA pipeline.
// GEMM staging uses GLOBAL_LOAD_ASYNC_TO_LDS (ASYNCcnt) with double buffering.
// ---------------------------------------------------------------------------

typedef __bf16 bf16_t;
typedef __attribute__((ext_vector_type(16))) __bf16 bf16x16;
typedef __attribute__((ext_vector_type(8)))  __bf16 bf16x8;
typedef __attribute__((ext_vector_type(4)))  __bf16 bf16x4;
typedef __attribute__((ext_vector_type(8)))  float  f32x8;
typedef __attribute__((ext_vector_type(4)))  float  f32x4;

#define B_    2
#define S_    2048
#define HID_  2048
#define HQ_   16
#define HKV_  8
#define DH_   128
#define MROWS (B_ * S_)                 // 4096
#define QKVN  ((HQ_ + 2 * HKV_) * DH_)  // 4096
#define SCALE 0.08838834764831845f      // 1/sqrt(128)

static __device__ __forceinline__ f32x8 wmma_bf16(bf16x16 a, bf16x16 b, f32x8 c) {
  return __builtin_amdgcn_wmma_f32_16x16x32_bf16(false, a, false, b, (short)0, c,
                                                 false, false);
}

static __device__ __forceinline__ bf16x16 mk16(bf16x8 lo, bf16x8 hi) {
  bf16x16 r;
#pragma unroll
  for (int i = 0; i < 8; ++i) { r[i] = lo[i]; r[i + 8] = hi[i]; }
  return r;
}

// Low 32 bits of a generic pointer into LDS are the LDS byte address
// (flat-LDS aperture maps addr[31:0] directly to LDS_ADDR).
static __device__ __forceinline__ unsigned lds_addr(const void* p) {
  return (unsigned)(unsigned long long)p;
}

// Async global->LDS 16-byte copy, tracked by ASYNCcnt.
static __device__ __forceinline__ void async_copy_b128(unsigned lds_byte_addr,
                                                       const void* gptr) {
  asm volatile("global_load_async_to_lds_b128 %0, %1, off"
               :: "v"(lds_byte_addr), "v"(gptr)
               : "memory");
}
static __device__ __forceinline__ void wait_async0() {
  asm volatile("s_wait_asynccnt 0" ::: "memory");
}

// ---------------------------------------------------------------------------
// Kernel 1: fp32 -> bf16 cast (n divisible by 4)
// ---------------------------------------------------------------------------
__global__ __launch_bounds__(256) void cast_bf16_kernel(
    const float* __restrict__ in, bf16_t* __restrict__ out, int n) {
  int i = (blockIdx.x * 256 + threadIdx.x) * 4;
  if (i < n) {
    f32x4 v = *(const f32x4*)(in + i);
    bf16x4 o;
    o[0] = (bf16_t)v.x; o[1] = (bf16_t)v.y; o[2] = (bf16_t)v.z; o[3] = (bf16_t)v.w;
    *(bf16x4*)(out + i) = o;
  }
}

// ---------------------------------------------------------------------------
// Kernel 2/5: C(MxN, f32) = A(MxK, bf16 row-major) * W(NxK, bf16 row-major)^T
// Block tile 128(M) x 256(N), K-step 32. 8 waves (2x4), wave tile 64x64
// (4x4 fragments -> 16 WMMA per k-step per wave, 16 DS fragment loads).
// Staging: async global->LDS, double-buffered.
// ---------------------------------------------------------------------------
#define GT_K   32
#define LDSTR  40     // halves; padded row stride (80B, 16B-aligned)
#define AR     128    // A rows per block
#define BR     256    // W rows (N cols) per block
#define TROWS  (AR + BR)  // 384 staged rows per k-step

__global__ __launch_bounds__(256) void gemm_bf16_kernel(
    const bf16_t* __restrict__ A, const bf16_t* __restrict__ W,
    float* __restrict__ C, int M, int N, int K) {
  __shared__ bf16_t sAB[2][TROWS * LDSTR];   // 2 x 30720B

  const int tid  = threadIdx.x;
  const int lane = tid & 31;
  const int wave = tid >> 5;     // 0..7
  const int wm   = wave >> 2;    // 0..1 : 64-row slab
  const int wn   = wave & 3;     // 0..3 : 64-col slab
  const int m0   = blockIdx.y * AR;
  const int n0   = blockIdx.x * BR;

  const int l16  = lane & 15;
  const int hi8  = (lane >> 4) * 8;
  const int hi16 = (lane >> 4) * 16;

  f32x8 acc[4][4];
#pragma unroll
  for (int i = 0; i < 4; ++i)
#pragma unroll
    for (int j = 0; j < 4; ++j) acc[i][j] = (f32x8)0.0f;

  const unsigned lbase0 = lds_addr(&sAB[0][0]);
  const unsigned lbase1 = lds_addr(&sAB[1][0]);

  // Stage one 384x32-half tile: 1536 16B chunks, 6 per thread, async to LDS.
  auto stage = [&](int buf, int k0) {
    unsigned base = buf ? lbase1 : lbase0;
#pragma unroll
    for (int c = 0; c < 6; ++c) {
      int idx = tid + c * 256;           // 0..1535
      int row = idx >> 2, seg = idx & 3; // seg: 8-half (16B) chunk
      const bf16_t* gp =
          (row < AR) ? A + (size_t)(m0 + row) * K + k0 + seg * 8
                     : W + (size_t)(n0 + (row - AR)) * K + k0 + seg * 8;
      async_copy_b128(base + (unsigned)(row * LDSTR + seg * 8) * 2, gp);
    }
  };

  stage(0, 0);
  wait_async0();
  __syncthreads();

  int cur = 0;
  for (int k0 = 0; k0 < K; k0 += GT_K) {
    if (k0 + GT_K < K) stage(cur ^ 1, k0 + GT_K);  // overlap next-tile DMA

    const bf16_t* sA = &sAB[cur][0];
    const bf16_t* sB = &sAB[cur][AR * LDSTR];

    bf16x16 af[4], bfr[4];
#pragma unroll
    for (int mf = 0; mf < 4; ++mf) {
      const bf16_t* ap = sA + (wm * 64 + mf * 16 + l16) * LDSTR;
      af[mf] = mk16(*(const bf16x8*)(ap + hi8),
                    *(const bf16x8*)(ap + 16 + hi8));
    }
#pragma unroll
    for (int nf = 0; nf < 4; ++nf) {
      const bf16_t* bp = sB + (wn * 64 + nf * 16 + l16) * LDSTR;
      bfr[nf] = mk16(*(const bf16x8*)(bp + hi8), *(const bf16x8*)(bp + 16 + hi8));
    }
#pragma unroll
    for (int mf = 0; mf < 4; ++mf)
#pragma unroll
      for (int nf = 0; nf < 4; ++nf)
        acc[mf][nf] = wmma_bf16(af[mf], bfr[nf], acc[mf][nf]);

    wait_async0();     // own async stores into next buffer are done
    __syncthreads();   // everyone done reading cur / writing next
    cur ^= 1;
  }

#pragma unroll
  for (int mf = 0; mf < 4; ++mf)
#pragma unroll
    for (int nf = 0; nf < 4; ++nf)
#pragma unroll
      for (int r = 0; r < 8; ++r) {
        int row = m0 + wm * 64 + mf * 16 + r + hi8;
        int col = n0 + wn * 64 + nf * 16 + l16;
        C[(size_t)row * N + col] = acc[mf][nf][r];
      }
}

// ---------------------------------------------------------------------------
// Kernel 3: per-head RMSNorm (q,k) + RoPE (q,k) + relayout/cast to bf16.
// ---------------------------------------------------------------------------
__global__ __launch_bounds__(256) void normrope_kernel(
    const float* __restrict__ qkv, const float* __restrict__ cosb,
    const float* __restrict__ sinb, const float* __restrict__ qw,
    const float* __restrict__ kw, bf16_t* __restrict__ Qb,
    bf16_t* __restrict__ Kb, bf16_t* __restrict__ Vb) {
  const int row  = blockIdx.x;          // b*S + s
  const int lane = threadIdx.x & 31;
  const int wave = threadIdx.x >> 5;
  const int dh   = lane * 4;
  const int b    = row >> 11;           // /S
  const int s    = row & (S_ - 1);

#pragma unroll
  for (int it = 0; it < 4; ++it) {
    const int head = wave + it * 8;     // 0..31
    int type, h, col0;
    if (head < HQ_)            { type = 0; h = head;        col0 = h * DH_; }
    else if (head < HQ_ + HKV_){ type = 1; h = head - HQ_;  col0 = HQ_ * DH_ + h * DH_; }
    else                       { type = 2; h = head - HQ_ - HKV_; col0 = (HQ_ + HKV_) * DH_ + h * DH_; }

    f32x4 x = *(const f32x4*)(qkv + (size_t)row * QKVN + col0 + dh);

    if (type < 2) {
      float ss = x.x * x.x + x.y * x.y + x.z * x.z + x.w * x.w;
#pragma unroll
      for (int m = 16; m >= 1; m >>= 1) ss += __shfl_xor(ss, m, 32);
      float rinv = rsqrtf(ss * (1.0f / DH_) + 1e-6f);
      const float* w = (type == 0) ? qw : kw;
      f32x4 wv = *(const f32x4*)(w + dh);
      x.x *= rinv * wv.x; x.y *= rinv * wv.y;
      x.z *= rinv * wv.z; x.w *= rinv * wv.w;

      // RoPE: element dh pairs with dh^64 -> partner lane = lane^16
      f32x4 xp;
      xp.x = __shfl_xor(x.x, 16, 32); xp.y = __shfl_xor(x.y, 16, 32);
      xp.z = __shfl_xor(x.z, 16, 32); xp.w = __shfl_xor(x.w, 16, 32);
      f32x4 cs = *(const f32x4*)(cosb + (size_t)row * DH_ + dh);
      f32x4 sn = *(const f32x4*)(sinb + (size_t)row * DH_ + dh);
      float sgn = (dh < DH_ / 2) ? -1.0f : 1.0f;
      x.x = x.x * cs.x + sgn * xp.x * sn.x;
      x.y = x.y * cs.y + sgn * xp.y * sn.y;
      x.z = x.z * cs.z + sgn * xp.z * sn.z;
      x.w = x.w * cs.w + sgn * xp.w * sn.w;
    }

    bf16x4 o;
    o[0] = (bf16_t)x.x; o[1] = (bf16_t)x.y; o[2] = (bf16_t)x.z; o[3] = (bf16_t)x.w;
    if (type == 0)
      *(bf16x4*)(Qb + (((size_t)b * HQ_ + h) * S_ + s) * DH_ + dh) = o;
    else if (type == 1)
      *(bf16x4*)(Kb + (((size_t)b * HKV_ + h) * S_ + s) * DH_ + dh) = o;
    else
      *(bf16x4*)(Vb + (((size_t)b * HKV_ + h) * S_ + s) * DH_ + dh) = o;
  }
}

// ---------------------------------------------------------------------------
// Kernel 4: flash attention (no mask). Block = 4 waves x 16 Q rows = 64 rows.
// Key tiles of 64. Scores: 16 WMMA / wave / tile; PV: 16 WMMA / wave / tile.
// ---------------------------------------------------------------------------
#define KT 64

__global__ __launch_bounds__(128) void attn_kernel(
    const bf16_t* __restrict__ Qb, const bf16_t* __restrict__ Kb,
    const bf16_t* __restrict__ Vb, bf16_t* __restrict__ Ob) {
  const int b    = blockIdx.z;
  const int hq   = blockIdx.y;
  const int hkv  = hq >> 1;  // G = 2
  const int q0   = blockIdx.x * 64;
  const int lane = threadIdx.x & 31;
  const int wave = threadIdx.x >> 5;   // 0..3
  const int l16  = lane & 15;
  const int hi8  = (lane >> 4) * 8;
  const int hi16 = (lane >> 4) * 16;

  const bf16_t* Qh = Qb + ((size_t)b * HQ_ + hq) * S_ * DH_;
  const bf16_t* Kh = Kb + ((size_t)b * HKV_ + hkv) * S_ * DH_;
  const bf16_t* Vh = Vb + ((size_t)b * HKV_ + hkv) * S_ * DH_;

  __shared__ bf16_t sVT[DH_][KT + 8];   // transposed V tile (18 KB)
  __shared__ bf16_t sP[4][16][KT];      // per-wave P tile  (8 KB)

  // Persistent Q A-fragments: 4 frags of K(dh)=32 for row l16 of this wave's tile
  const int qrow = q0 + wave * 16 + l16;
  bf16x16 qf[4];
#pragma unroll
  for (int kf = 0; kf < 4; ++kf) {
    const bf16_t* qp = Qh + (size_t)qrow * DH_ + kf * 32;
    qf[kf] = mk16(*(const bf16x8*)(qp + hi8), *(const bf16x8*)(qp + 16 + hi8));
  }

  f32x8 oacc[8];
#pragma unroll
  for (int t = 0; t < 8; ++t) oacc[t] = (f32x8)0.0f;
  float mrow[8], lrow[8];
#pragma unroll
  for (int r = 0; r < 8; ++r) { mrow[r] = -3.0e38f; lrow[r] = 0.0f; }

  for (int kt = 0; kt < S_; kt += KT) {
    // Stage V tile transposed: sVT[dh][key]
    for (int sgi = threadIdx.x; sgi < KT * 16; sgi += 128) {
      int key = sgi >> 4, seg = sgi & 15;
      bf16x8 v = *(const bf16x8*)(Vh + (size_t)(kt + key) * DH_ + seg * 8);
#pragma unroll
      for (int j = 0; j < 8; ++j) sVT[seg * 8 + j][key] = v[j];
    }
    __syncthreads();

    // Scores: Q(16x128) @ K_tile^T(128x64) -> 4 column tiles
    f32x8 sc[4];
#pragma unroll
    for (int nt = 0; nt < 4; ++nt) {
      f32x8 a = (f32x8)0.0f;
#pragma unroll
      for (int kf = 0; kf < 4; ++kf) {
        const bf16_t* kp =
            Kh + (size_t)(kt + nt * 16 + l16) * DH_ + kf * 32;
        bf16x16 bfr = mk16(*(const bf16x8*)(kp + hi8), *(const bf16x8*)(kp + 16 + hi8));
        a = wmma_bf16(qf[kf], bfr, a);
      }
      sc[nt] = a;
    }

    // Online softmax (rows spread across 16 lanes; 8 rows per lane in VGPRs)
#pragma unroll
    for (int nt = 0; nt < 4; ++nt)
#pragma unroll
      for (int r = 0; r < 8; ++r) sc[nt][r] *= SCALE;

#pragma unroll
    for (int r = 0; r < 8; ++r) {
      float mx = fmaxf(fmaxf(sc[0][r], sc[1][r]), fmaxf(sc[2][r], sc[3][r]));
#pragma unroll
      for (int d = 8; d >= 1; d >>= 1) mx = fmaxf(mx, __shfl_xor(mx, d, 32));
      float nm    = fmaxf(mrow[r], mx);
      float alpha = __expf(mrow[r] - nm);
      mrow[r] = nm;
      float rs = 0.0f;
#pragma unroll
      for (int nt = 0; nt < 4; ++nt) {
        float p = __expf(sc[nt][r] - nm);
        rs += p;
        sP[wave][r + hi8][nt * 16 + l16] = (bf16_t)p;
      }
#pragma unroll
      for (int d = 8; d >= 1; d >>= 1) rs += __shfl_xor(rs, d, 32);
      lrow[r] = lrow[r] * alpha + rs;
#pragma unroll
      for (int t = 0; t < 8; ++t) oacc[t][r] *= alpha;
    }

    // Ensure P stores are complete before re-reading (same-wave DS in-order)
    asm volatile("s_wait_dscnt 0" ::: "memory");

    // O += P(16x64) @ V(64x128)
#pragma unroll
    for (int nt = 0; nt < 8; ++nt) {
#pragma unroll
      for (int kf = 0; kf < 2; ++kf) {
        const bf16_t* pp = &sP[wave][l16][kf * 32];
        bf16x16 af = mk16(*(const bf16x8*)(pp + hi8),
                          *(const bf16x8*)(pp + 16 + hi8));
        const bf16_t* vp = &sVT[nt * 16 + l16][kf * 32];
        bf16x16 bfr = mk16(*(const bf16x8*)(vp + hi8), *(const bf16x8*)(vp + 16 + hi8));
        oacc[nt] = wmma_bf16(af, bfr, oacc[nt]);
      }
    }
    __syncthreads();
  }

  // Epilogue: write O/l to [b][s][hq*DH + dh] (bf16, A-matrix for O-proj GEMM)
#pragma unroll
  for (int nt = 0; nt < 8; ++nt)
#pragma unroll
    for (int r = 0; r < 8; ++r) {
      int m    = r + hi8;
      int srow = q0 + wave * 16 + m;
      float v  = oacc[nt][r] / lrow[r];
      Ob[((size_t)b * S_ + srow) * HID_ + hq * DH_ + nt * 16 + l16] = (bf16_t)v;
    }
}

// ---------------------------------------------------------------------------
// Host-side launch
// ---------------------------------------------------------------------------
extern "C" void kernel_launch(void* const* d_in, const int* in_sizes, int n_in,
                              void* d_out, int out_size, void* d_ws,
                              size_t ws_size, hipStream_t stream) {
  const float* hidden = (const float*)d_in[0];
  const float* cosb   = (const float*)d_in[1];
  const float* sinb   = (const float*)d_in[2];
  const float* qkv_w  = (const float*)d_in[3];
  const float* q_nw   = (const float*)d_in[4];
  const float* k_nw   = (const float*)d_in[5];
  const float* o_w    = (const float*)d_in[6];
  float* out = (float*)d_out;

  char*  ws  = (char*)d_ws;
  size_t off = 0;
  auto carve = [&](size_t bytes) -> char* {
    char* p = ws + off;
    off += (bytes + 255) & ~(size_t)255;
    return p;
  };
  bf16_t* Xb    = (bf16_t*)carve((size_t)MROWS * HID_ * 2);
  bf16_t* Wqkvb = (bf16_t*)carve((size_t)QKVN * HID_ * 2);
  bf16_t* Wob   = (bf16_t*)carve((size_t)HID_ * HID_ * 2);
  float*  QKV   = (float*)carve((size_t)MROWS * QKVN * 4);
  bf16_t* Qb    = (bf16_t*)carve((size_t)B_ * HQ_ * S_ * DH_ * 2);
  bf16_t* Kb    = (bf16_t*)carve((size_t)B_ * HKV_ * S_ * DH_ * 2);
  bf16_t* Vb    = (bf16_t*)carve((size_t)B_ * HKV_ * S_ * DH_ * 2);
  bf16_t* Ab    = (bf16_t*)carve((size_t)MROWS * HID_ * 2);

  // 1) fp32 -> bf16 casts
  {
    int n = MROWS * HID_;
    cast_bf16_kernel<<<n / 1024, 256, 0, stream>>>(hidden, Xb, n);
    n = QKVN * HID_;
    cast_bf16_kernel<<<n / 1024, 256, 0, stream>>>(qkv_w, Wqkvb, n);
    n = HID_ * HID_;
    cast_bf16_kernel<<<n / 1024, 256, 0, stream>>>(o_w, Wob, n);
  }

  // 2) QKV projection: [4096,4096] = X[4096,2048] @ Wqkv^T
  gemm_bf16_kernel<<<dim3(QKVN / BR, MROWS / AR), 256, 0, stream>>>(
      Xb, Wqkvb, QKV, MROWS, QKVN, HID_);

  // 3) RMSNorm + RoPE + relayout
  normrope_kernel<<<MROWS, 256, 0, stream>>>(QKV, cosb, sinb, q_nw, k_nw, Qb,
                                             Kb, Vb);

  // 4) Attention
  attn_kernel<<<dim3(S_ / 64, HQ_, B_), 128, 0, stream>>>(Qb, Kb, Vb, Ab);

  // 5) Output projection: out[4096,2048] = A[4096,2048] @ Wo^T
  gemm_bf16_kernel<<<dim3(HID_ / BR, MROWS / AR), 256, 0, stream>>>(
      Ab, Wob, out, MROWS, HID_, HID_);
}
